// QuantWindowAttention_6442450944082
// MI455X (gfx1250) — matrix operand order repaired and checked by reading, not compiled, not run on hardware
//
#include <hip/hip_runtime.h>
#include <hip/hip_bf16.h>

// CDNA5 / gfx1250 fused Swin window attention.
// One workgroup (256 threads = 8 wave32) per window. All intermediates in LDS
// (~224 KB of the 320 KB WGP budget). All GEMMs use v_wmma_f32_16x16x32_bf16.
// Fragment gathers are 16-byte contiguous runs -> ds_load_b128; bf16
// conversion uses the native hardware convert via __bf16 casts.

typedef __attribute__((ext_vector_type(16))) __bf16 v16bf;
typedef __attribute__((ext_vector_type(8)))  float  v8f;

union FragB   { v16bf v; unsigned short u[16]; unsigned int d[8]; };
union FragAcc { v8f v; float f[8]; };
union Pack8   { uint4 q; unsigned short s[8]; __bf16 h[8]; };
union Pack4   { uint2 q; unsigned short s[4]; __bf16 h[4]; };

__device__ __forceinline__ unsigned short f2bf(float f) {
  union { __bf16 h; unsigned short s; } u;
  u.h = (__bf16)f;                 // native v_cvt bf16 (RNE)
  return u.s;
}

// A-matrix fragment (16-bit, 16x32): lane holds K=kb..kb+7 and K=kb+16..kb+23
// -> two 16-byte contiguous LDS loads. p points at &row[kb].
__device__ __forceinline__ FragB load_frag_gap(const unsigned short* p) {
  FragB f;
  uint4 a = *(const uint4*)p;          // 8 bf16
  uint4 b = *(const uint4*)(p + 16);   // 8 bf16, 32B further
  f.d[0] = a.x; f.d[1] = a.y; f.d[2] = a.z; f.d[3] = a.w;
  f.d[4] = b.x; f.d[5] = b.y; f.d[6] = b.z; f.d[7] = b.w;
  return f;
}

// B-matrix fragment: lane holds K=kb..kb+7 and K=kb+16..kb+23 (same as A).
__device__ __forceinline__ FragB load_frag_cont(const unsigned short* p) {
  FragB f;
  uint4 a = *(const uint4*)p;
  uint4 b = *(const uint4*)(p + 16);
  f.d[0] = a.x; f.d[1] = a.y; f.d[2] = a.z; f.d[3] = a.w;
  f.d[4] = b.x; f.d[5] = b.y; f.d[6] = b.z; f.d[7] = b.w;
  return f;
}

// ---------------------------------------------------------------------------
// Prep: f32 weights -> bf16 in workspace; bias[h][i][j] = table[rel_index[ij], h]
// ---------------------------------------------------------------------------
__global__ void swin_prep_kernel(const float* __restrict__ qkv_w,
                                 const float* __restrict__ proj_w,
                                 const float* __restrict__ table,
                                 const int*   __restrict__ rel_index,
                                 unsigned short* __restrict__ qkv_bf,
                                 unsigned short* __restrict__ proj_bf,
                                 float* __restrict__ bias) {
  int idx = blockIdx.x * blockDim.x + threadIdx.x;
  if (idx < 196608) {
    qkv_bf[idx] = f2bf(qkv_w[idx]);
  } else if (idx < 262144) {
    int t = idx - 196608;
    proj_bf[t] = f2bf(proj_w[t]);
  } else if (idx < 294912) {
    int t = idx - 262144;            // t = h*4096 + i*64 + j
    int h  = t >> 12;
    int ij = t & 4095;
    bias[t] = table[rel_index[ij] * 8 + h];
  }
}

// ---------------------------------------------------------------------------
// Fused window attention kernel
// ---------------------------------------------------------------------------
__global__ __launch_bounds__(256, 1)
void swin_attn_kernel(const float* __restrict__ x,
                      const float* __restrict__ mask,
                      const unsigned short* __restrict__ qkv_wbf,
                      const float* __restrict__ qkv_b,
                      const unsigned short* __restrict__ proj_wbf,
                      const float* __restrict__ proj_b,
                      const float* __restrict__ bias,
                      float* __restrict__ out) {
  extern __shared__ unsigned short smem[];
  unsigned short* Xb = smem;                  // [64][256]   bf16 32 KB
  unsigned short* Qs = Xb + 64 * 256;         // [8][64][32] bf16 32 KB (pre-scaled, tok-major)
  unsigned short* Ks = Qs + 8 * 64 * 32;      // [8][64][32] bf16 32 KB (tok-major)
  unsigned short* Vt = Ks + 8 * 64 * 32;      // [8][32][64] bf16 32 KB (dim-major: V^T)
  unsigned short* Ps = Vt + 8 * 64 * 32;      // [8][64][64] bf16 64 KB (softmax probs)
  unsigned short* Os = Ps + 8 * 64 * 64;      // [64][256]   bf16 32 KB ([tok][h*32+d])

  const int b   = blockIdx.x;
  const int tid = threadIdx.x;
  const int w   = tid >> 5;    // wave 0..7
  const int l   = tid & 31;    // lane 0..31
  const int ln  = l & 15;      // lane within half
  const int hhi = l >> 4;      // lane half

  // ---- S1: stage x[b] (64x256 f32) -> bf16 LDS (b64 packed stores) ------
  const float* xb = x + (size_t)b * (64 * 256);
  for (int i = tid; i < 4096; i += 256) {     // float4 per iter
    float4 v = ((const float4*)xb)[i];
    Pack4 pk;
    pk.h[0] = (__bf16)v.x; pk.h[1] = (__bf16)v.y;
    pk.h[2] = (__bf16)v.z; pk.h[3] = (__bf16)v.w;
    *(uint2*)&Xb[i * 4] = pk.q;
  }
  __syncthreads();

  // ---- S2: qkv = x @ qkv_w^T + qkv_b  (M=64, N=768, K=256) --------------
  const float scale = 0.17677669529663687f;   // hd^-0.5
  for (int mt = 0; mt < 4; ++mt) {
    // cache A fragments (all 8 k-steps) for this 16-row block
    FragB A[8];
    {
      const unsigned short* row = &Xb[(mt * 16 + ln) * 256];
      #pragma unroll
      for (int ks = 0; ks < 8; ++ks)
        A[ks] = load_frag_gap(row + ks * 32 + hhi * 8);
    }
    for (int nt = w; nt < 48; nt += 8) {
      FragAcc acc;
      #pragma unroll
      for (int r = 0; r < 8; ++r) acc.f[r] = 0.0f;
      int ncol = nt * 16 + ln;
      const unsigned short* wrow = &qkv_wbf[ncol * 256];   // B[k][n] = W[n][k]
      #pragma unroll
      for (int ks = 0; ks < 8; ++ks) {
        FragB Bf = load_frag_cont(wrow + ks * 32 + hhi * 8);
        acc.v = __builtin_amdgcn_wmma_f32_16x16x32_bf16(
            false, A[ks].v, false, Bf.v, (short)0, acc.v, false, false);
      }
      // bias, split into per-head Q (scaled) / K / V^T buffers
      float bb  = qkv_b[ncol];
      int part  = ncol >> 8;           // 0=q 1=k 2=v
      int h     = (ncol >> 5) & 7;
      int d     = ncol & 31;
      if (part == 2) {
        // V transposed: rows r are consecutive tokens -> one b128 store
        Pack8 pk;
        #pragma unroll
        for (int r = 0; r < 8; ++r) pk.h[r] = (__bf16)(acc.f[r] + bb);
        *(uint4*)&Vt[h * 2048 + d * 64 + mt * 16 + hhi * 8] = pk.q;
      } else {
        unsigned short* dst = (part == 0) ? Qs : Ks;
        float sc = (part == 0) ? scale : 1.0f;
        #pragma unroll
        for (int r = 0; r < 8; ++r) {
          int mrow = mt * 16 + hhi * 8 + r;
          dst[h * 2048 + mrow * 32 + d] = f2bf((acc.f[r] + bb) * sc);
        }
      }
    }
  }
  __syncthreads();

  // ---- S3: attention, wave w <-> head w ---------------------------------
  {
    const int h = w;
    const unsigned short* Qh = &Qs[h * 2048];
    const unsigned short* Kh = &Ks[h * 2048];
    const unsigned short* Vh = &Vt[h * 2048];
    unsigned short*       Ph = &Ps[h * 4096];
    const float* biash = &bias[h * 4096];
    const float* maskw = &mask[(size_t)(b & 63) * 4096];

    // B fragments from K: B[d][j] = K[j][d], contiguous d per row
    FragB BK[4];
    #pragma unroll
    for (int nt = 0; nt < 4; ++nt)
      BK[nt] = load_frag_cont(&Kh[(nt * 16 + ln) * 32 + hhi * 8]);

    for (int mt = 0; mt < 4; ++mt) {
      // A fragment from Q (K=32 -> single WMMA per tile)
      FragB AQ = load_frag_gap(&Qh[(mt * 16 + ln) * 32 + hhi * 8]);
      FragAcc acc[4];
      #pragma unroll
      for (int nt = 0; nt < 4; ++nt) {
        #pragma unroll
        for (int r = 0; r < 8; ++r) acc[nt].f[r] = 0.0f;
        acc[nt].v = __builtin_amdgcn_wmma_f32_16x16x32_bf16(
            false, AQ.v, false, BK[nt].v, (short)0, acc[nt].v, false, false);
      }
      // + relative-position bias + shifted-window mask (L2-resident)
      #pragma unroll
      for (int nt = 0; nt < 4; ++nt) {
        int j = nt * 16 + ln;
        #pragma unroll
        for (int r = 0; r < 8; ++r) {
          int i = mt * 16 + hhi * 8 + r;
          acc[nt].f[r] += biash[i * 64 + j] + maskw[i * 64 + j];
        }
      }
      // row softmax: row lives in one lane-half (16 lanes) x 4 column tiles
      #pragma unroll
      for (int r = 0; r < 8; ++r) {
        float rmax = fmaxf(fmaxf(acc[0].f[r], acc[1].f[r]),
                           fmaxf(acc[2].f[r], acc[3].f[r]));
        #pragma unroll
        for (int off = 1; off < 16; off <<= 1)
          rmax = fmaxf(rmax, __shfl_xor(rmax, off, 32));
        float e0 = __expf(acc[0].f[r] - rmax);
        float e1 = __expf(acc[1].f[r] - rmax);
        float e2 = __expf(acc[2].f[r] - rmax);
        float e3 = __expf(acc[3].f[r] - rmax);
        float rsum = e0 + e1 + e2 + e3;
        #pragma unroll
        for (int off = 1; off < 16; off <<= 1)
          rsum += __shfl_xor(rsum, off, 32);
        float inv = 1.0f / rsum;
        acc[0].f[r] = e0 * inv; acc[1].f[r] = e1 * inv;
        acc[2].f[r] = e2 * inv; acc[3].f[r] = e3 * inv;
      }
      // store probabilities as bf16 A-matrix for P @ V
      #pragma unroll
      for (int nt = 0; nt < 4; ++nt) {
        int j = nt * 16 + ln;
        #pragma unroll
        for (int r = 0; r < 8; ++r) {
          int i = mt * 16 + hhi * 8 + r;
          Ph[i * 64 + j] = f2bf(acc[nt].f[r]);
        }
      }
    }

    // out_h = P @ V  (M=64, N=32, K=64); B from V^T has contiguous K
    for (int mt = 0; mt < 4; ++mt) {
      FragB AP[2];
      {
        const unsigned short* row = &Ph[(mt * 16 + ln) * 64];
        #pragma unroll
        for (int ks = 0; ks < 2; ++ks)
          AP[ks] = load_frag_gap(row + ks * 32 + hhi * 8);
      }
      #pragma unroll
      for (int dt = 0; dt < 2; ++dt) {
        FragAcc acc2;
        #pragma unroll
        for (int r = 0; r < 8; ++r) acc2.f[r] = 0.0f;
        int d = dt * 16 + ln;
        const unsigned short* vrow = &Vh[d * 64];   // V^T row: tokens contiguous
        #pragma unroll
        for (int ks = 0; ks < 2; ++ks) {
          FragB BV = load_frag_cont(vrow + ks * 32 + hhi * 8);
          acc2.v = __builtin_amdgcn_wmma_f32_16x16x32_bf16(
              false, AP[ks].v, false, BV.v, (short)0, acc2.v, false, false);
        }
        #pragma unroll
        for (int r = 0; r < 8; ++r) {
          int mrow = mt * 16 + hhi * 8 + r;
          Os[mrow * 256 + h * 32 + d] = f2bf(acc2.f[r]);
        }
      }
    }
  }
  __syncthreads();

  // ---- S4: out = O @ proj_w^T + proj_b  (M=64, N=256, K=256) ------------
  float* outb = out + (size_t)b * (64 * 256);
  for (int t = w; t < 64; t += 8) {
    int mt = t >> 4, nt = t & 15;
    int ncol = nt * 16 + ln;
    FragAcc acc;
    #pragma unroll
    for (int r = 0; r < 8; ++r) acc.f[r] = 0.0f;
    const unsigned short* arow = &Os[(mt * 16 + ln) * 256];
    const unsigned short* wrow = &proj_wbf[ncol * 256];
    #pragma unroll
    for (int ks = 0; ks < 8; ++ks) {
      FragB Af = load_frag_gap(arow + ks * 32 + hhi * 8);
      FragB Bf = load_frag_cont(wrow + ks * 32 + hhi * 8);
      acc.v = __builtin_amdgcn_wmma_f32_16x16x32_bf16(
          false, Af.v, false, Bf.v, (short)0, acc.v, false, false);
    }
    float bb = proj_b[ncol];
    #pragma unroll
    for (int r = 0; r < 8; ++r) {
      int mrow = mt * 16 + hhi * 8 + r;
      outb[mrow * 256 + ncol] = acc.f[r] + bb;
    }
  }
}

// ---------------------------------------------------------------------------
extern "C" void kernel_launch(void* const* d_in, const int* in_sizes, int n_in,
                              void* d_out, int out_size, void* d_ws, size_t ws_size,
                              hipStream_t stream) {
  const float* x      = (const float*)d_in[0];
  const float* mask   = (const float*)d_in[1];
  const float* qkv_w  = (const float*)d_in[2];
  const float* qkv_b  = (const float*)d_in[3];
  const float* proj_w = (const float*)d_in[4];
  const float* proj_b = (const float*)d_in[5];
  const float* table  = (const float*)d_in[6];
  const int*   relidx = (const int*)d_in[7];

  // workspace layout: qkv_w bf16 (384 KB) | proj_w bf16 (128 KB) | bias f32 (128 KB)
  unsigned short* qkv_bf  = (unsigned short*)d_ws;
  unsigned short* proj_bf = qkv_bf + 196608;
  float*          bias    = (float*)(proj_bf + 65536);

  swin_prep_kernel<<<1152, 256, 0, stream>>>(qkv_w, proj_w, table, relidx,
                                             qkv_bf, proj_bf, bias);

  size_t lds_bytes = (size_t)(64 * 256           // Xb
                              + 3 * 8 * 64 * 32  // Q, K, V^T
                              + 8 * 64 * 64      // P
                              + 64 * 256) * 2;   // O   => 229376 B (< 320 KB WGP)
  swin_attn_kernel<<<4096, 256, lds_bytes, stream>>>(
      x, mask, qkv_bf, (const float*)qkv_b, proj_bf, (const float*)proj_b,
      bias, (float*)d_out);
}
